// ConcatAttention_3925600108938
// MI455X (gfx1250) — matrix operand order repaired and checked by reading, not compiled, not run on hardware
//
#include <hip/hip_runtime.h>
#include <hip/hip_bf16.h>
#include <math.h>

// ---------------------------------------------------------------------------
// Problem dims (fixed by reference): B=64, S=2048, A=AD=QD=1024
// out = [weightedContext 64*1024][score 64*2048][precompute 64*2048*1024]
// ---------------------------------------------------------------------------
#define BATCH 64
#define SEQ   2048
#define DIM   1024          // A == AD == QD
#define MROWS (BATCH * SEQ) // 131072 GEMM rows

typedef __attribute__((ext_vector_type(16))) __bf16 v16bf;
typedef __attribute__((ext_vector_type(8)))  float  v8f;

// ---------------------------------------------------------------------------
// Kernel 1: split W_pre (f32 [A=1024, K=1024], row-major, B[n,k]) into bf16
// hi/lo fragments laid out in WMMA B-operand order:
//   fragment(nt, kt): lane L, element e ->  n = nt*16 + (L&15)
//                                           k = kt*32 + 16*(L>>4) + e
// flat index f = ((nt*32 + kt)*32 + lane)*16 + e
// ---------------------------------------------------------------------------
__global__ void prep_wpre_kernel(const float* __restrict__ Wp,
                                 __bf16* __restrict__ bhi,
                                 __bf16* __restrict__ blo) {
    int idx = blockIdx.x * 256 + threadIdx.x;      // 0 .. 1048575
    int e    = idx & 15;
    int lane = (idx >> 4) & 31;
    int kt   = (idx >> 9) & 31;
    int nt   = idx >> 14;                          // 0..63
    int n = nt * 16 + (lane & 15);
    int k = kt * 32 + 16 * (lane >> 4) + e;
    float x = Wp[(size_t)n * DIM + k];
    __bf16 h = (__bf16)x;
    bhi[idx] = h;
    blo[idx] = (__bf16)(x - (float)h);
}

// ---------------------------------------------------------------------------
// Kernel 2: targetT[b][a] = sum_k input[b][k] * W_q[a][k]
// grid (4, 64), block 256: b = blockIdx.y, a = blockIdx.x*256 + tid
// ---------------------------------------------------------------------------
__global__ void targetT_kernel(const float* __restrict__ input,
                               const float* __restrict__ Wq,
                               float* __restrict__ tT) {
    __shared__ float inrow[DIM];
    int t = threadIdx.x;
    int b = blockIdx.y;
    int a = blockIdx.x * 256 + t;
    #pragma unroll
    for (int r = 0; r < 4; ++r)
        inrow[r * 256 + t] = input[(size_t)b * DIM + r * 256 + t];
    __syncthreads();
    const float4* w = (const float4*)(Wq + (size_t)a * DIM);
    float acc = 0.0f;
    #pragma unroll 4
    for (int k4 = 0; k4 < DIM / 4; ++k4) {
        float4 wv = w[k4];
        acc += wv.x * inrow[4 * k4 + 0];
        acc += wv.y * inrow[4 * k4 + 1];
        acc += wv.z * inrow[4 * k4 + 2];
        acc += wv.w * inrow[4 * k4 + 3];
    }
    tT[(size_t)b * DIM + a] = acc;
}

// ---------------------------------------------------------------------------
// Kernel 3: main GEMM  precompute = context @ W_pre^T + b_pre   (WMMA bf16x3)
// fused epilogue: energy[m] += sum_n v[n]*tanh(c[m,n] + targetT[b,n])
// grid (8192, 4), block 128 (4 waves). Wave w covers n-tiles
//   blockIdx.y*16 + w*4 .. +3   (4 tiles of 16 cols = 64 cols per wave)
// ---------------------------------------------------------------------------
__global__ void __launch_bounds__(128)
gemm_pre_energy_kernel(const float* __restrict__ ctx,
                       const float* __restrict__ b_pre,
                       const float* __restrict__ vvec,
                       const __bf16* __restrict__ bhi,
                       const __bf16* __restrict__ blo,
                       const float* __restrict__ tT,
                       float* __restrict__ energy,
                       float* __restrict__ out_pre) {
    const int lane = threadIdx.x & 31;
    const int wave = threadIdx.x >> 5;
    const int mt   = blockIdx.x;                        // row tile 0..8191
    const int nt0  = blockIdx.y * 16 + wave * 4;        // first n-tile of 4
    const int hs   = lane >> 4;
    const int l15  = lane & 15;

    const int mrow = mt * 16 + l15;                     // A-matrix row for lane
    const float* arow = ctx + (size_t)mrow * DIM;

    v8f acc[4];
    #pragma unroll
    for (int j = 0; j < 4; ++j) acc[j] = (v8f)0.0f;

    const v16bf* fhi = (const v16bf*)bhi;
    const v16bf* flo = (const v16bf*)blo;

    for (int kt = 0; kt < 32; ++kt) {
        // ---- load A tile (fp32) in contiguous k order (matches B fragments), split hi/lo
        const float* c0 = arow + kt * 32 + 16 * hs;      // K = 16*hs + 0..7
        const float* c1 = c0 + 8;                        // K = 16*hs + 8..15
        float4 f0 = ((const float4*)c0)[0];
        float4 f1 = ((const float4*)c0)[1];
        float4 f2 = ((const float4*)c1)[0];
        float4 f3 = ((const float4*)c1)[1];
        if (kt < 31) __builtin_prefetch(arow + (kt + 1) * 32 + 8 * hs, 0, 1);

        float vals[16] = {f0.x, f0.y, f0.z, f0.w,  f1.x, f1.y, f1.z, f1.w,
                          f2.x, f2.y, f2.z, f2.w,  f3.x, f3.y, f3.z, f3.w};
        v16bf ahi, alo;
        #pragma unroll
        for (int e = 0; e < 16; ++e) {
            __bf16 h = (__bf16)vals[e];
            ahi[e] = h;
            alo[e] = (__bf16)(vals[e] - (float)h);
        }

        // ---- 4 n-tiles x 3 WMMAs (hi*hi + hi*lo + lo*hi)
        #pragma unroll
        for (int j = 0; j < 4; ++j) {
            int frag = ((nt0 + j) * 32 + kt) * 32 + lane;
            v16bf wh = fhi[frag];
            v16bf wl = flo[frag];
            acc[j] = __builtin_amdgcn_wmma_f32_16x16x32_bf16(
                false, ahi, false, wh, (short)0, acc[j], false, false);
            acc[j] = __builtin_amdgcn_wmma_f32_16x16x32_bf16(
                false, ahi, false, wl, (short)0, acc[j], false, false);
            acc[j] = __builtin_amdgcn_wmma_f32_16x16x32_bf16(
                false, alo, false, wh, (short)0, acc[j], false, false);
        }
    }

    // ---- epilogue: +b_pre, store precompute, fused energy partials
    const int b = (mt * 16) >> 11;                      // batch of this tile
    float esum[8];
    #pragma unroll
    for (int i = 0; i < 8; ++i) esum[i] = 0.0f;

    #pragma unroll
    for (int j = 0; j < 4; ++j) {
        int n = (nt0 + j) * 16 + l15;                   // column for this lane
        float bp = b_pre[n];
        float ta = tT[(size_t)b * DIM + n];
        float vv = vvec[n];
        #pragma unroll
        for (int i = 0; i < 8; ++i) {
            float c = acc[j][i] + bp;                   // precompute value
            int m = mt * 16 + i + 8 * hs;               // C layout: row = i+8*hs
            out_pre[(size_t)m * DIM + n] = c;
            esum[i] += vv * tanhf(c + ta);
        }
    }

    // reduce across the 16 lanes of each half (rows i+0 in lanes 0-15,
    // rows i+8 in lanes 16-31)
    #pragma unroll
    for (int off = 1; off < 16; off <<= 1) {
        #pragma unroll
        for (int i = 0; i < 8; ++i)
            esum[i] += __shfl_xor(esum[i], off, 32);
    }
    if (l15 == 0) {
        #pragma unroll
        for (int i = 0; i < 8; ++i)
            atomicAdd(&energy[mt * 16 + i + 8 * hs], esum[i]);
    }
}

// ---------------------------------------------------------------------------
// Kernel 4a: softmax over S per batch. grid 64, block 256 (8 elems/thread)
// ---------------------------------------------------------------------------
__global__ void softmax_kernel(const float* __restrict__ energy,
                               float* __restrict__ score) {
    __shared__ float red[256];
    int t = threadIdx.x;
    int b = blockIdx.x;
    float e[8];
    float lmax = -1e30f;
    #pragma unroll
    for (int r = 0; r < 8; ++r) {
        e[r] = energy[(size_t)b * SEQ + r * 256 + t];
        lmax = fmaxf(lmax, e[r]);
    }
    red[t] = lmax; __syncthreads();
    for (int s = 128; s > 0; s >>= 1) {
        if (t < s) red[t] = fmaxf(red[t], red[t + s]);
        __syncthreads();
    }
    float m = red[0]; __syncthreads();
    float lsum = 0.0f;
    #pragma unroll
    for (int r = 0; r < 8; ++r) { e[r] = expf(e[r] - m); lsum += e[r]; }
    red[t] = lsum; __syncthreads();
    for (int s = 128; s > 0; s >>= 1) {
        if (t < s) red[t] += red[t + s];
        __syncthreads();
    }
    float inv = 1.0f / red[0];
    #pragma unroll
    for (int r = 0; r < 8; ++r)
        score[(size_t)b * SEQ + r * 256 + t] = e[r] * inv;
}

// ---------------------------------------------------------------------------
// Kernel 4b: weightedContext[b][d] = sum_s score[b][s]*context[b][s][d]
// grid (4, 64), block 256: d = blockIdx.x*256 + t
// ---------------------------------------------------------------------------
__global__ void weighted_kernel(const float* __restrict__ score,
                                const float* __restrict__ ctx,
                                float* __restrict__ out_wc) {
    __shared__ float sc[SEQ];
    int t = threadIdx.x;
    int b = blockIdx.y;
    int d = blockIdx.x * 256 + t;
    #pragma unroll
    for (int r = 0; r < 8; ++r)
        sc[r * 256 + t] = score[(size_t)b * SEQ + r * 256 + t];
    __syncthreads();
    const float* cb = ctx + (size_t)b * SEQ * DIM + d;
    float acc = 0.0f;
    #pragma unroll 4
    for (int s = 0; s < SEQ; ++s)
        acc = fmaf(sc[s], cb[(size_t)s * DIM], acc);
    out_wc[(size_t)b * DIM + d] = acc;
}

// ---------------------------------------------------------------------------
// launcher
// ---------------------------------------------------------------------------
extern "C" void kernel_launch(void* const* d_in, const int* in_sizes, int n_in,
                              void* d_out, int out_size, void* d_ws, size_t ws_size,
                              hipStream_t stream) {
    const float* input  = (const float*)d_in[0];   // [64,1024]
    const float* ctx    = (const float*)d_in[1];   // [64,2048,1024]
    const float* W_pre  = (const float*)d_in[2];   // [1024,1024]
    const float* b_pre  = (const float*)d_in[3];   // [1024]
    const float* W_q    = (const float*)d_in[4];   // [1024,1024]
    const float* vvec   = (const float*)d_in[5];   // [1024]

    float* out_wc    = (float*)d_out;                               // 65536
    float* out_score = out_wc + (size_t)BATCH * DIM;                // 131072
    float* out_pre   = out_score + (size_t)BATCH * SEQ;             // 134217728

    // workspace layout
    __bf16* wsBhi = (__bf16*)d_ws;                                  // 2 MB
    __bf16* wsBlo = wsBhi + (1 << 20);                              // 2 MB
    float*  tT     = (float*)((char*)d_ws + 4u * 1024 * 1024);      // 256 KB
    float*  energy = tT + (size_t)BATCH * DIM;                      // 512 KB

    // 1) W_pre -> bf16 hi/lo fragments
    prep_wpre_kernel<<<4096, 256, 0, stream>>>(W_pre, wsBhi, wsBlo);

    // 2) targetT = input @ W_q^T
    targetT_kernel<<<dim3(4, 64), 256, 0, stream>>>(input, W_q, tT);

    // 3) zero energy accumulator (graph-capture-safe async memset)
    hipMemsetAsync(energy, 0, (size_t)BATCH * SEQ * sizeof(float), stream);

    // 4) main WMMA GEMM + fused tanh/energy epilogue
    gemm_pre_energy_kernel<<<dim3(MROWS / 16, 4), 128, 0, stream>>>(
        ctx, b_pre, vvec, wsBhi, wsBlo, tT, energy, out_pre);

    // 5) softmax -> score output
    softmax_kernel<<<BATCH, 256, 0, stream>>>(energy, out_score);

    // 6) weightedContext
    weighted_kernel<<<dim3(4, BATCH), 256, 0, stream>>>(out_score, ctx, out_wc);
}
